// SwinTransformerBlock_75660143886356
// MI455X (gfx1250) — compile-verified, matrix operand order checked by reading, not run on hardware
//
#include <hip/hip_runtime.h>
#include <hip/hip_bf16.h>

typedef __bf16 bf16;
typedef __attribute__((ext_vector_type(16))) __bf16 v16bf;
typedef __attribute__((ext_vector_type(8)))  float  v8f;
typedef unsigned int u32;
typedef unsigned long long u64;
typedef __attribute__((ext_vector_type(4))) u32 v4u;
typedef __attribute__((ext_vector_type(8))) int v8i;
typedef __attribute__((ext_vector_type(4))) int v4i;

#define CCH   192      // channels
#define NQKV  576      // 3*C
#define HID   768      // 4*C
#define NTOK  128      // tokens per window (2*8*8)
#define NWIN  512      // total windows (B * 4*8*8)
#define NHEAD 6
#define MROWS 65536    // NWIN*NTOK == B*T*H*W tokens

// ---------------------------------------------------------------- WMMA utils
static __device__ __forceinline__ v8f wmma_bf16(v16bf a, v16bf b, v8f c) {
  return __builtin_amdgcn_wmma_f32_16x16x32_bf16(false, a, false, b, (short)0, c,
                                                 false, false);
}

// A/B fragment from row-major [16 x 32] tile (ISA 7.12.2 16-bit A layout).
static __device__ __forceinline__ v16bf frag_ld(const bf16* __restrict__ p0,
                                                int ld, int lane) {
  const int r  = lane & 15;
  const int kb = (lane & 16) ? 8 : 0;
  const bf16* p = p0 + (size_t)r * ld + kb;
  v16bf f;
#pragma unroll
  for (int e = 0; e < 8; ++e) { f[e] = p[e]; f[e + 8] = p[e + 16]; }
  return f;
}

// ---------------------------------------------------------------- TDM
// LDS byte offset of a __shared__ object: low 32 bits of the flat address
// (LDS aperture keeps the offset in addr[31:0], ISA §10.2).
static __device__ __forceinline__ u32 lofs(const void* p) {
  return (u32)(u64)p;
}

// TENSOR_LOAD_TO_LDS of a 2D bf16 tile: tile_dim0=32 contiguous elements,
// tile_dim1=rows, row stride = stride_elems. D# per ISA §8.3/8.4.
static __device__ __forceinline__ void tdm_load_2d(const void* gsrc, u32 lds_off,
                                                   u32 rows, u32 stride_elems) {
  const u64 ga = (u64)gsrc;
  v4u g0;
  g0[0] = 1u;                                            // count=1, user mode
  g0[1] = lds_off;                                       // lds_addr
  g0[2] = (u32)ga;                                       // global_addr[31:0]
  g0[3] = (u32)((ga >> 32) & 0x01FFFFFFu) | 0x80000000u; // ga[56:32] | type=2
  const u32 td0 = 32u;                                   // tensor_dim0 == tile_dim0
  const u32 td1 = rows;
  const u64 s0  = stride_elems;                          // tensor_dim0_stride
  v8i g1;
  g1[0] = (int)(1u << 16);                               // data_size=1 (2 bytes)
  g1[1] = (int)(td0 << 16);                              // tensor_dim0[15:0]
  g1[2] = (int)((td0 >> 16) | (td1 << 16));              // td0[31:16] | td1[15:0]
  g1[3] = (int)((td1 >> 16) | (32u << 16));              // td1[31:16] | tile_dim0=32
  g1[4] = (int)(rows & 0xFFFFu);                         // tile_dim1=rows, tile_dim2=0
  g1[5] = (int)(u32)(s0 & 0xFFFFFFFFu);                  // stride[31:0]
  g1[6] = (int)(u32)((s0 >> 32) & 0xFFFFu);              // stride[47:32], dim1_stride=0
  g1[7] = 0;
  const v4i g2 = {0, 0, 0, 0};                           // dims 2/3 unused
  const v4i g3 = {0, 0, 0, 0};                           // dim 4 unused
  asm volatile("tensor_load_to_lds %0, %1, %2, %3"
               :: "s"(g0), "s"(g1), "s"(g2), "s"(g3)
               : "memory");
}

// ---------------------------------------------------------------- GEMM core
// 128x64 output tile per block; A/W tiles staged to LDS by the TDM, double
// buffered; 8 waves compute 16x64 strips out of LDS.
static __device__ __forceinline__ void gemm_tdm(const bf16* __restrict__ A,
                                                const bf16* __restrict__ W,
                                                int K, int rowBlock, int colBlock,
                                                int wv, int lane,
                                                bf16* lA, bf16* lW, v8f acc[4]) {
  if (wv == 0) {
    tdm_load_2d(A + (size_t)rowBlock * K, lofs(lA), 128u, (u32)K);
    tdm_load_2d(W + (size_t)colBlock * K, lofs(lW), 64u, (u32)K);
  }
  int cur = 0;
  for (int kt = 0; kt < K; kt += 32) {
    if (wv == 0) {
      if (kt + 32 < K) {
        tdm_load_2d(A + (size_t)rowBlock * K + kt + 32,
                    lofs(lA + (cur ^ 1) * (128 * 32)), 128u, (u32)K);
        tdm_load_2d(W + (size_t)colBlock * K + kt + 32,
                    lofs(lW + (cur ^ 1) * (64 * 32)), 64u, (u32)K);
        __builtin_amdgcn_s_wait_tensorcnt(2);  // current pair landed
      } else {
        __builtin_amdgcn_s_wait_tensorcnt(0);
      }
    }
    __syncthreads();
    v16bf af = frag_ld(lA + cur * (128 * 32) + wv * 16 * 32, 32, lane);
#pragma unroll
    for (int t = 0; t < 4; ++t) {
      v16bf wf = frag_ld(lW + cur * (64 * 32) + t * 16 * 32, 32, lane);
      acc[t] = wmma_bf16(af, wf, acc[t]);
    }
    __syncthreads();   // protect buffer before TDM overwrites it next iter
    cur ^= 1;
  }
}

// ---------------------------------------------------------------- helpers
__global__ void k_f2bf(const float* __restrict__ s, bf16* __restrict__ d, int n) {
  int i = blockIdx.x * blockDim.x + threadIdx.x;
  if (i < n) d[i] = (bf16)s[i];
}

// Precompute rel-pos bias per head: biasf[h][n][m] = rpb_table[rpb_index[n,m], h]
__global__ void k_bias(const float* __restrict__ rpb_table,
                       const int* __restrict__ rpb_index,
                       float* __restrict__ biasf) {
  int i = blockIdx.x * blockDim.x + threadIdx.x;
  if (i < NHEAD * NTOK * NTOK) {
    const int h  = i / (NTOK * NTOK);
    const int nm = i % (NTOK * NTOK);
    biasf[i] = rpb_table[(size_t)rpb_index[nm] * NHEAD + h];
  }
}

// roll(-1,-4,-4) + window partition, fp32 -> bf16. One block per windowed row.
__global__ void k_winpart(const float* __restrict__ x, bf16* __restrict__ xw) {
  const int r = blockIdx.x;       // windowed token row (b*128 + n)
  const int c = threadIdx.x;      // channel
  const int b = r >> 7, n = r & 127;
  const int batch = b >> 8, rem = b & 255;
  const int t = (((rem >> 6) * 2 + (n >> 6)) + 1) & 7;
  const int h = ((((rem >> 3) & 7) * 8 + ((n >> 3) & 7)) + 4) & 63;
  const int w = (((rem & 7) * 8 + (n & 7)) + 4) & 63;
  const size_t tok = (((size_t)batch * 8 + t) * 64 + h) * 64 + w;
  xw[(size_t)r * CCH + c] = (bf16)x[tok * CCH + c];
}

// ---------------------------------------------------------------- GEMMs
__global__ __launch_bounds__(256) void k_gemm_qkv(const bf16* __restrict__ A,
    const bf16* __restrict__ W, const float* __restrict__ bias,
    bf16* __restrict__ C) {
  __shared__ bf16 lA[2][128 * 32];
  __shared__ bf16 lW[2][64 * 32];
  const int lane = threadIdx.x & 31, wv = threadIdx.x >> 5;
  const int rowBlock = blockIdx.y * 128, colBlock = blockIdx.x * 64;
  const v8f vz = {0.f,0.f,0.f,0.f,0.f,0.f,0.f,0.f};
  v8f acc[4] = {vz, vz, vz, vz};
  gemm_tdm(A, W, CCH, rowBlock, colBlock, wv, lane, &lA[0][0], &lW[0][0], acc);
  const int ro = (lane & 16) ? 8 : 0, cl = lane & 15;
#pragma unroll
  for (int t = 0; t < 4; ++t)
#pragma unroll
    for (int i = 0; i < 8; ++i) {
      const int r = rowBlock + wv * 16 + ro + i;
      const int c = colBlock + t * 16 + cl;
      C[(size_t)r * NQKV + c] = (bf16)(acc[t][i] + bias[c]);
    }
}

// proj: fuses bias + window-reverse + roll(+1,+4,+4) + residual (y=z+x),
// writes y as bf16 in natural spatial order for the MLP.
__global__ __launch_bounds__(256) void k_gemm_proj(const bf16* __restrict__ A,
    const bf16* __restrict__ W, const float* __restrict__ bias,
    const float* __restrict__ x, bf16* __restrict__ ybf) {
  __shared__ bf16 lA[2][128 * 32];
  __shared__ bf16 lW[2][64 * 32];
  const int lane = threadIdx.x & 31, wv = threadIdx.x >> 5;
  const int rowBlock = blockIdx.y * 128, colBlock = blockIdx.x * 64;
  const v8f vz = {0.f,0.f,0.f,0.f,0.f,0.f,0.f,0.f};
  v8f acc[4] = {vz, vz, vz, vz};
  gemm_tdm(A, W, CCH, rowBlock, colBlock, wv, lane, &lA[0][0], &lW[0][0], acc);
  const int ro = (lane & 16) ? 8 : 0, cl = lane & 15;
#pragma unroll
  for (int t = 0; t < 4; ++t)
#pragma unroll
    for (int i = 0; i < 8; ++i) {
      const int r = rowBlock + wv * 16 + ro + i;
      const int c = colBlock + t * 16 + cl;
      const int b = r >> 7, n = r & 127;
      const int batch = b >> 8, rem = b & 255;
      const int tt = (((rem >> 6) * 2 + (n >> 6)) + 1) & 7;
      const int hh = ((((rem >> 3) & 7) * 8 + ((n >> 3) & 7)) + 4) & 63;
      const int ww = (((rem & 7) * 8 + (n & 7)) + 4) & 63;
      const size_t tok = (((size_t)batch * 8 + tt) * 64 + hh) * 64 + ww;
      const size_t o = tok * CCH + c;
      ybf[o] = (bf16)(acc[t][i] + bias[c] + x[o]);
    }
}

__global__ __launch_bounds__(256) void k_gemm_fc1(const bf16* __restrict__ A,
    const bf16* __restrict__ W, const float* __restrict__ bias,
    bf16* __restrict__ H) {
  __shared__ bf16 lA[2][128 * 32];
  __shared__ bf16 lW[2][64 * 32];
  const int lane = threadIdx.x & 31, wv = threadIdx.x >> 5;
  const int rowBlock = blockIdx.y * 128, colBlock = blockIdx.x * 64;
  const v8f vz = {0.f,0.f,0.f,0.f,0.f,0.f,0.f,0.f};
  v8f acc[4] = {vz, vz, vz, vz};
  gemm_tdm(A, W, CCH, rowBlock, colBlock, wv, lane, &lA[0][0], &lW[0][0], acc);
  const int ro = (lane & 16) ? 8 : 0, cl = lane & 15;
#pragma unroll
  for (int t = 0; t < 4; ++t)
#pragma unroll
    for (int i = 0; i < 8; ++i) {
      const int r = rowBlock + wv * 16 + ro + i;
      const int c = colBlock + t * 16 + cl;
      const float v = acc[t][i] + bias[c];
      const float g = 0.5f * v * (1.0f + erff(v * 0.7071067811865475f));
      H[(size_t)r * HID + c] = (bf16)g;
    }
}

// fc2: fuses bias + final residual (out = x + mlp), fp32 output.
__global__ __launch_bounds__(256) void k_gemm_fc2(const bf16* __restrict__ A,
    const bf16* __restrict__ W, const float* __restrict__ bias,
    const float* __restrict__ x, float* __restrict__ outp) {
  __shared__ bf16 lA[2][128 * 32];
  __shared__ bf16 lW[2][64 * 32];
  const int lane = threadIdx.x & 31, wv = threadIdx.x >> 5;
  const int rowBlock = blockIdx.y * 128, colBlock = blockIdx.x * 64;
  const v8f vz = {0.f,0.f,0.f,0.f,0.f,0.f,0.f,0.f};
  v8f acc[4] = {vz, vz, vz, vz};
  gemm_tdm(A, W, HID, rowBlock, colBlock, wv, lane, &lA[0][0], &lW[0][0], acc);
  const int ro = (lane & 16) ? 8 : 0, cl = lane & 15;
#pragma unroll
  for (int t = 0; t < 4; ++t)
#pragma unroll
    for (int i = 0; i < 8; ++i) {
      const int r = rowBlock + wv * 16 + ro + i;
      const int c = colBlock + t * 16 + cl;
      const size_t o = (size_t)r * CCH + c;
      outp[o] = acc[t][i] + bias[c] + x[o];
    }
}

// ---------------------------------------------------------------- attention
// One block per (window, head): TDM-staged Q/K/V -> QK^T -> scale + bias +
// shift mask -> softmax (register/shuffle) -> transpose V once -> P@V.
__global__ __launch_bounds__(256) void k_attn(const bf16* __restrict__ qkv,
                                              const float* __restrict__ biasf,
                                              const float* __restrict__ mask,
                                              bf16* __restrict__ outp) {
  __shared__ bf16 q_s[NTOK * 32];
  __shared__ bf16 k_s[NTOK * 32];
  __shared__ bf16 v_s[NTOK * 32];
  __shared__ bf16 vT_s[32 * NTOK];     // V transposed: [d][token]
  __shared__ bf16 p_s[NTOK * NTOK];
  const int b = blockIdx.x;
  const int head = blockIdx.y;
  const int tid = threadIdx.x;
  const int lane = tid & 31;
  const int wv = tid >> 5;

  if (wv == 0) {
    const bf16* qp = qkv + (size_t)b * NTOK * NQKV + head * 32;
    tdm_load_2d(qp,           lofs(q_s), NTOK, NQKV);  // Q  (stride 576)
    tdm_load_2d(qp + CCH,     lofs(k_s), NTOK, NQKV);  // K
    tdm_load_2d(qp + 2 * CCH, lofs(v_s), NTOK, NQKV);  // V
    __builtin_amdgcn_s_wait_tensorcnt(0);
  }
  __syncthreads();

  const v8f vz = {0.f,0.f,0.f,0.f,0.f,0.f,0.f,0.f};
  v8f acc[8];
#pragma unroll
  for (int t = 0; t < 8; ++t) acc[t] = vz;

  // S = Q K^T : wave wv owns rows [16wv,16wv+16), d=32 -> single WMMA per tile
  const v16bf qf = frag_ld(q_s + wv * 16 * 32, 32, lane);
#pragma unroll
  for (int t = 0; t < 8; ++t) {
    v16bf kf = frag_ld(k_s + t * 16 * 32, 32, lane);
    acc[t] = wmma_bf16(qf, kf, acc[t]);
  }

  const float scale = 0.1767766952966369f;   // 1/sqrt(32)
  const int ro = (lane & 16) ? 8 : 0;
  const int cl = lane & 15;
  const int wIdx = b & 255;                  // window index within batch image
  const float* bh = biasf + (size_t)head * NTOK * NTOK;
  const float* mk = mask + (size_t)wIdx * NTOK * NTOK;

#pragma unroll
  for (int i = 0; i < 8; ++i) {
    const int n = wv * 16 + ro + i;
    float rm = -3.0e38f;
#pragma unroll
    for (int t = 0; t < 8; ++t) {
      const int m = t * 16 + cl;
      float s = acc[t][i] * scale + bh[n * NTOK + m] + mk[n * NTOK + m];
      acc[t][i] = s;
      rm = fmaxf(rm, s);
    }
    // row reductions stay inside the 16-lane group that co-owns this row
    for (int off = 1; off < 16; off <<= 1) rm = fmaxf(rm, __shfl_xor(rm, off, 32));
    float sum = 0.f;
#pragma unroll
    for (int t = 0; t < 8; ++t) {
      const float e = __expf(acc[t][i] - rm);
      acc[t][i] = e;
      sum += e;
    }
    for (int off = 1; off < 16; off <<= 1) sum += __shfl_xor(sum, off, 32);
    const float inv = 1.0f / sum;
#pragma unroll
    for (int t = 0; t < 8; ++t)
      p_s[n * NTOK + t * 16 + cl] = (bf16)(acc[t][i] * inv);
  }

  // transpose V once: vT[d][m] = v[m][d] (v_s ready since the TDM barrier)
  for (int i = tid; i < 32 * NTOK; i += 256) {
    const int dd = i >> 7, m = i & 127;
    vT_s[i] = v_s[m * 32 + dd];
  }
  __syncthreads();

  // O = P V : rows [16wv,16wv+16), cols 0..31, K=128; B-frags contiguous in vT
  v8f oa[2] = {vz, vz};
  for (int kt = 0; kt < NTOK; kt += 32) {
    v16bf pf = frag_ld(p_s + wv * 16 * NTOK + kt, NTOK, lane);
#pragma unroll
    for (int t2 = 0; t2 < 2; ++t2) {
      v16bf vf = frag_ld(vT_s + (size_t)(t2 * 16) * NTOK + kt, NTOK, lane);
      oa[t2] = wmma_bf16(pf, vf, oa[t2]);
    }
  }
#pragma unroll
  for (int t2 = 0; t2 < 2; ++t2)
#pragma unroll
    for (int i = 0; i < 8; ++i) {
      const int n = wv * 16 + ro + i;
      const int c = head * 32 + t2 * 16 + cl;
      outp[((size_t)b * NTOK + n) * CCH + c] = (bf16)oa[t2][i];
    }
}

// ---------------------------------------------------------------- launcher
extern "C" void kernel_launch(void* const* d_in, const int* in_sizes, int n_in,
                              void* d_out, int out_size, void* d_ws, size_t ws_size,
                              hipStream_t stream) {
  (void)in_sizes; (void)n_in; (void)out_size; (void)ws_size;
  const float* x         = (const float*)d_in[0];
  const float* attn_mask = (const float*)d_in[1];
  const float* qkv_w     = (const float*)d_in[2];
  const float* qkv_b     = (const float*)d_in[3];
  const float* proj_w    = (const float*)d_in[4];
  const float* proj_b    = (const float*)d_in[5];
  const float* rpb_table = (const float*)d_in[6];
  const int*   rpb_index = (const int*)d_in[7];
  const float* fc1_w     = (const float*)d_in[8];
  const float* fc1_b     = (const float*)d_in[9];
  const float* fc2_w     = (const float*)d_in[10];
  const float* fc2_b     = (const float*)d_in[11];
  float* outp = (float*)d_out;

  char* ws = (char*)d_ws;
  size_t off = 0;
  bf16* w_qkv  = (bf16*)(ws + off); off += (size_t)NQKV * CCH * 2;
  bf16* w_proj = (bf16*)(ws + off); off += (size_t)CCH * CCH * 2;
  bf16* w_fc1  = (bf16*)(ws + off); off += (size_t)HID * CCH * 2;
  bf16* w_fc2  = (bf16*)(ws + off); off += (size_t)CCH * HID * 2;
  off = (off + 255) & ~(size_t)255;
  float* biasf = (float*)(ws + off); off += (size_t)NHEAD * NTOK * NTOK * 4;
  bf16* xw     = (bf16*)(ws + off); off += (size_t)MROWS * CCH * 2;   // reused
  bf16* attnO  = xw;                                                  // as attn out
  bf16* qkvbuf = (bf16*)(ws + off); off += (size_t)MROWS * NQKV * 2;
  bf16* ybf    = (bf16*)(ws + off); off += (size_t)MROWS * CCH * 2;
  bf16* hidbuf = (bf16*)(ws + off); off += (size_t)MROWS * HID * 2;

  // weight conversion fp32 -> bf16
  k_f2bf<<<dim3((NQKV * CCH + 255) / 256), dim3(256), 0, stream>>>(qkv_w, w_qkv, NQKV * CCH);
  k_f2bf<<<dim3((CCH * CCH + 255) / 256), dim3(256), 0, stream>>>(proj_w, w_proj, CCH * CCH);
  k_f2bf<<<dim3((HID * CCH + 255) / 256), dim3(256), 0, stream>>>(fc1_w, w_fc1, HID * CCH);
  k_f2bf<<<dim3((CCH * HID + 255) / 256), dim3(256), 0, stream>>>(fc2_w, w_fc2, CCH * HID);

  // precompute per-head rel-pos bias table (breaks the dependent gather chain)
  k_bias<<<dim3((NHEAD * NTOK * NTOK + 255) / 256), dim3(256), 0, stream>>>(
      rpb_table, rpb_index, biasf);

  // shift + window partition
  k_winpart<<<dim3(MROWS), dim3(CCH), 0, stream>>>(x, xw);

  // QKV projection
  k_gemm_qkv<<<dim3(NQKV / 64, MROWS / 128), dim3(256), 0, stream>>>(xw, w_qkv, qkv_b, qkvbuf);

  // fused windowed attention per (window, head)
  k_attn<<<dim3(NWIN, NHEAD), dim3(256), 0, stream>>>(qkvbuf, biasf, attn_mask, attnO);

  // output projection + window reverse + roll back + residual
  k_gemm_proj<<<dim3(CCH / 64, MROWS / 128), dim3(256), 0, stream>>>(attnO, w_proj,
                                                                     proj_b, x, ybf);
  // MLP
  k_gemm_fc1<<<dim3(HID / 64, MROWS / 128), dim3(256), 0, stream>>>(ybf, w_fc1, fc1_b, hidbuf);
  k_gemm_fc2<<<dim3(CCH / 64, MROWS / 128), dim3(256), 0, stream>>>(hidbuf, w_fc2, fc2_b, x, outp);
}
